// DecoderMoEModel_31078383354371
// MI455X (gfx1250) — compile-verified, matrix operand order checked by reading, not run on hardware
//
#include <hip/hip_runtime.h>
#include <hip/hip_bf16.h>
#include <math.h>
#include <stdint.h>

// ---- model constants (match reference) ----
constexpr int cV = 32000, cD = 512, cH = 8, cL = 4;
constexpr int cHID = 2048, cE = 8;
constexpr int cT = 2048, cHD = 64;
constexpr int SLOT_CAP   = cT * 2 + cE * 64;   // 4608 padded routing slots
constexpr int SLOT_TILES = SLOT_CAP / 64;      // 72 GEMM m-tiles
constexpr float cEPS = 1e-6f;

typedef __attribute__((ext_vector_type(16))) __bf16 v16bf;
typedef __attribute__((ext_vector_type(8)))  __bf16 bf16x8;
typedef __attribute__((ext_vector_type(8)))  float  v8f;

static __device__ __forceinline__ bf16x8 pack8(float4 a, float4 b) {
  bf16x8 r;
  r[0]=(__bf16)a.x; r[1]=(__bf16)a.y; r[2]=(__bf16)a.z; r[3]=(__bf16)a.w;
  r[4]=(__bf16)b.x; r[5]=(__bf16)b.y; r[6]=(__bf16)b.z; r[7]=(__bf16)b.w;
  return r;
}
static __device__ __forceinline__ v16bf cat8(bf16x8 lo, bf16x8 hi) {
  return __builtin_shufflevector(lo, hi, 0,1,2,3,4,5,6,7,8,9,10,11,12,13,14,15);
}

// ============================================================================
// Generic bf16 WMMA GEMM: C[M,N] = A[M,K] @ B[K,N]   (fp32 in/out, bf16 MACs)
//  - 64x64 tile per 256-thread block (8 wave32s, each wave -> 16x32 outputs)
//  - GATHER: A row indices come from rowIdx (MoE token gather)
//  - TRANSB: B is stored [N,K] row-major (logits GEMM vs tok_embed)
//  - ACC:    C += result (residual adds)
//  - tileExpert: per-m-tile expert id selects B += e*expertStride (MoE)
// ============================================================================
template<bool GATHER, bool TRANSB, bool ACC>
__global__ __launch_bounds__(256) void k_gemm(
    const float* __restrict__ A, const float* __restrict__ B,
    float* __restrict__ C, int N, int K,
    const int* __restrict__ rowIdx,
    const int* __restrict__ tileExpert, size_t expertStride,
    const int* __restrict__ dTotalTiles)
{
  const int tm = blockIdx.x, tn = blockIdx.y;
  if (dTotalTiles && tm >= *dTotalTiles) return;         // skip empty expert tiles
  if (tileExpert) B += (size_t)tileExpert[tm] * expertStride;

  __shared__ __bf16 As[64][40];   // 64 rows x 32 k (+8 pad), 80B stride (16B aligned)
  __shared__ __bf16 Bs[64][40];   // stored n-major: Bs[n][k]

  const int tid  = threadIdx.x;
  const int lane = tid & 31;
  const int wv   = tid >> 5;
  const int wm   = wv & 3, wn = wv >> 2;
  const int half = lane >> 4, idx = lane & 15;

  v8f acc0 = {0,0,0,0,0,0,0,0};
  v8f acc1 = {0,0,0,0,0,0,0,0};

  // A global-load assignment: 64 rows x 32 k, 8 floats per thread
  const int arow = tid >> 2, kseg = (tid & 3) * 8;
  size_t aRowOff;
  {
    int gr = tm * 64 + arow;
    int sr = GATHER ? rowIdx[gr] : gr;
    aRowOff = (size_t)sr * K;
  }
  // B global-load assignment
  const int bi0 = TRANSB ? (tid >> 2) : (tid >> 3);        // n-row | k-row
  const int bi1 = TRANSB ? (tid & 3) * 8 : (tid & 7) * 8;  // kseg  | nseg

  const int frow = wm * 16 + idx;   // A fragment row within tile
  const int fcol = wn * 32 + idx;   // B fragment col (first of two 16-wide tiles)

  for (int kt = 0; kt < K; kt += 32) {
    const float* ap = A + aRowOff + kt + kseg;
    float4 a0 = *(const float4*)ap;
    float4 a1 = *(const float4*)(ap + 4);

    const float* bp = TRANSB
        ? B + (size_t)(tn * 64 + bi0) * K + kt + bi1
        : B + (size_t)(kt + bi0) * N + tn * 64 + bi1;
    float4 b0 = *(const float4*)bp;
    float4 b1 = *(const float4*)(bp + 4);

    if (kt + 32 < K) {                       // gfx1250 global_prefetch_b8
      __builtin_prefetch(ap + 32, 0, 3);
      __builtin_prefetch(TRANSB ? bp + 32 : bp + (size_t)32 * N, 0, 3);
    }

    __syncthreads();   // previous compute done before LDS overwrite
    *(bf16x8*)&As[arow][kseg] = pack8(a0, a1);
    if (TRANSB) {
      *(bf16x8*)&Bs[bi0][bi1] = pack8(b0, b1);
    } else {           // transpose fp32 [k][n] -> bf16 Bs[n][k]
      const float bb[8] = {b0.x,b0.y,b0.z,b0.w,b1.x,b1.y,b1.z,b1.w};
      #pragma unroll
      for (int i = 0; i < 8; ++i) Bs[bi1 + i][bi0] = (__bf16)bb[i];
    }
    __syncthreads();

    // fragments per ISA 7.12.2: lane<16 -> K{0..7,16..23}, lane>=16 -> K{8..15,24..31}
    bf16x8 al = *(const bf16x8*)&As[frow][half * 8];
    bf16x8 ah = *(const bf16x8*)&As[frow][16 + half * 8];
    v16bf  af = cat8(al, ah);

    bf16x8 bl0 = *(const bf16x8*)&Bs[fcol][half * 8];
    bf16x8 bh0 = *(const bf16x8*)&Bs[fcol][16 + half * 8];
    v16bf  bf0 = cat8(bl0, bh0);
    acc0 = __builtin_amdgcn_wmma_f32_16x16x32_bf16(false, af, false, bf0,
                                                   (short)0, acc0, false, false);
    bf16x8 bl1 = *(const bf16x8*)&Bs[fcol + 16][half * 8];
    bf16x8 bh1 = *(const bf16x8*)&Bs[fcol + 16][16 + half * 8];
    v16bf  bf1 = cat8(bl1, bh1);
    acc1 = __builtin_amdgcn_wmma_f32_16x16x32_bf16(false, af, false, bf1,
                                                   (short)0, acc1, false, false);
  }

  // C/D layout: lanes 0-15 hold M=r, lanes 16-31 hold M=r+8
  const int rbase = tm * 64 + wm * 16 + half * 8;
  const int cbase = tn * 64 + wn * 32 + idx;
  #pragma unroll
  for (int r = 0; r < 8; ++r) {
    size_t o = (size_t)(rbase + r) * N + cbase;
    if (ACC) { C[o] += acc0[r]; C[o + 16] += acc1[r]; }
    else     { C[o]  = acc0[r]; C[o + 16]  = acc1[r]; }
  }
}

// ============================================================================
// Elementwise / attention / routing kernels
// ============================================================================
__global__ void k_embed(const int* __restrict__ ids, const float* __restrict__ emb,
                        float* __restrict__ x) {
  int t = blockIdx.x, tok = ids[t];
  for (int d = threadIdx.x; d < cD; d += 256)
    x[(size_t)t * cD + d] = emb[(size_t)tok * cD + d];
}

__global__ __launch_bounds__(256) void k_rmsnorm(const float* __restrict__ x,
                                                 const float* __restrict__ w,
                                                 float* __restrict__ o) {
  const int t = blockIdx.x, tid = threadIdx.x;
  const float* xr = x + (size_t)t * cD;
  float v0 = xr[tid], v1 = xr[tid + 256];
  float s = v0 * v0 + v1 * v1;
  for (int off = 16; off; off >>= 1) s += __shfl_xor(s, off, 32);
  __shared__ float red[8];
  __shared__ float rinv;
  if ((tid & 31) == 0) red[tid >> 5] = s;
  __syncthreads();
  if (tid == 0) {
    float tot = 0.f;
    for (int i = 0; i < 8; ++i) tot += red[i];
    rinv = rsqrtf(tot / (float)cD + cEPS);
  }
  __syncthreads();
  float r = rinv;
  o[(size_t)t * cD + tid]       = v0 * r * w[tid];
  o[(size_t)t * cD + tid + 256] = v1 * r * w[tid + 256];
}

__global__ __launch_bounds__(256) void k_rope(float* __restrict__ q, float* __restrict__ k) {
  const int t = blockIdx.x, j = threadIdx.x;
  const int h = j >> 5, i = j & 31;
  float inv = __powf(10000.f, -(2.f * (float)i) / (float)cHD);
  float ang = (float)t * inv;
  float c = cosf(ang), s = sinf(ang);
  size_t b = ((size_t)t * cH + h) * cHD;
  float q1 = q[b + i], q2 = q[b + i + 32];
  q[b + i]      = q1 * c - q2 * s;
  q[b + i + 32] = q2 * c + q1 * s;
  float k1 = k[b + i], k2 = k[b + i + 32];
  k[b + i]      = k1 * c - k2 * s;
  k[b + i + 32] = k2 * c + k1 * s;
}

// causal online-softmax attention; one wave32 per (token, head).
// K/V tiles are staged into LDS with gfx1250 ASYNC global->LDS DMA
// (global_load_async_to_lds_b128 + s_wait_asynccnt): fp32 bytes are copied
// verbatim, so the async engine replaces the VGPR round-trip entirely.
__global__ __launch_bounds__(256) void k_attn(const float* __restrict__ q,
                                              const float* __restrict__ k,
                                              const float* __restrict__ v,
                                              float* __restrict__ o) {
  __shared__ float Ks[64][68];   // 272B row stride: 16B aligned
  __shared__ float Vs[64][68];
  const int h = blockIdx.y;
  const int lane = threadIdx.x, wy = threadIdx.y;
  const int t = blockIdx.x * 8 + wy;
  const int tmax = blockIdx.x * 8 + 7;
  const size_t qb = ((size_t)t * cH + h) * cHD;
  const float qa = q[qb + lane] * 0.125f;
  const float qc = q[qb + lane + 32] * 0.125f;
  float m = -3.0e38f, lsum = 0.f, a0 = 0.f, a1 = 0.f;
  const int tid = wy * 32 + lane;
  const int lrow = tid >> 2, lseg = (tid & 3) * 16;
  // wave-relative LDS byte offsets (low 32 bits of the flat shared address)
  const uint32_t ldsK = (uint32_t)(uintptr_t)&Ks[lrow][lseg];
  const uint32_t ldsV = (uint32_t)(uintptr_t)&Vs[lrow][lseg];
  for (int sb = 0; sb <= tmax; sb += 64) {
    __syncthreads();   // previous tile fully consumed before async overwrite
    size_t gb = ((size_t)(sb + lrow) * cH + h) * cHD + lseg;
    const float* gk = &k[gb];
    const float* gv = &v[gb];
    #pragma unroll
    for (int i = 0; i < 4; ++i) {   // 16 floats per thread per array, 16B chunks
      asm volatile("global_load_async_to_lds_b128 %0, %1, off"
                   :: "v"(ldsK + 16u * i), "v"(gk + 4 * i) : "memory");
      asm volatile("global_load_async_to_lds_b128 %0, %1, off"
                   :: "v"(ldsV + 16u * i), "v"(gv + 4 * i) : "memory");
    }
    asm volatile("s_wait_asynccnt 0x0" ::: "memory");  // own DMA landed in LDS
    __syncthreads();                                   // everyone's DMA landed
    int jmax = min(63, t - sb);
    for (int j = 0; j <= jmax; ++j) {
      float d = qa * Ks[j][lane] + qc * Ks[j][lane + 32];
      for (int off = 16; off; off >>= 1) d += __shfl_xor(d, off, 32);
      float mn = fmaxf(m, d);
      float f = __expf(m - mn), p = __expf(d - mn);
      lsum = lsum * f + p;
      a0 = a0 * f + p * Vs[j][lane];
      a1 = a1 * f + p * Vs[j][lane + 32];
      m = mn;
    }
  }
  o[qb + lane]      = a0 / lsum;
  o[qb + lane + 32] = a1 / lsum;
}

// per-token router: logits (E=8), top-2, softmax weights, expert counts
__global__ __launch_bounds__(256) void k_router(const float* __restrict__ xn,
                                                const float* __restrict__ rw,
                                                int* __restrict__ tokExp,
                                                float* __restrict__ tokW,
                                                int* __restrict__ count) {
  const int lane = threadIdx.x;
  const int t = blockIdx.x * 8 + threadIdx.y;
  float lg[8] = {0,0,0,0,0,0,0,0};
  const float* xr = xn + (size_t)t * cD;
  for (int d = lane; d < cD; d += 32) {
    float xv = xr[d];
    const float* rwd = rw + (size_t)d * cE;
    #pragma unroll
    for (int e = 0; e < 8; ++e) lg[e] += xv * rwd[e];
  }
  #pragma unroll
  for (int e = 0; e < 8; ++e)
    for (int off = 16; off; off >>= 1) lg[e] += __shfl_xor(lg[e], off, 32);
  if (lane == 0) {
    int e0 = 0; float v0 = lg[0];
    for (int e = 1; e < 8; ++e) if (lg[e] > v0) { v0 = lg[e]; e0 = e; }
    int e1 = (e0 == 0) ? 1 : 0; float v1 = lg[e1];
    for (int e = 0; e < 8; ++e) if (e != e0 && lg[e] > v1) { v1 = lg[e]; e1 = e; }
    float w1e = __expf(v1 - v0);
    float s = 1.f + w1e;
    tokExp[2*t] = e0;  tokExp[2*t+1] = e1;
    tokW[2*t] = 1.f / s; tokW[2*t+1] = w1e / s;
    atomicAdd(&count[e0], 1);
    atomicAdd(&count[e1], 1);
  }
}

__global__ void k_zero8(int* count) { if (threadIdx.x < 8) count[threadIdx.x] = 0; }

// single block: 64-aligned per-expert segment offsets, tile->expert map, slot init
__global__ __launch_bounds__(256) void k_scan_init(const int* __restrict__ count,
                                                   int* padOff, int* cursor,
                                                   int* slot2token, int* tileExpert,
                                                   int* totalTiles) {
  __shared__ int sOff[9];
  if (threadIdx.x == 0) {
    int tot = 0;
    for (int e = 0; e < 8; ++e) {
      sOff[e] = tot; padOff[e] = tot; cursor[e] = 0;
      tot += (count[e] + 63) & ~63;
    }
    sOff[8] = tot; padOff[8] = tot;
    *totalTiles = tot / 64;
  }
  __syncthreads();
  for (int i = threadIdx.x; i < SLOT_CAP; i += 256) slot2token[i] = 0;
  for (int i = threadIdx.x; i < SLOT_TILES; i += 256) {
    int slot = i * 64, e = 0;
    for (int j = 0; j < 8; ++j) if (slot >= sOff[j] && slot < sOff[j + 1]) e = j;
    tileExpert[i] = e;
  }
}

__global__ void k_assign(const int* __restrict__ tokExp, const int* __restrict__ padOff,
                         int* cursor, int* slot2token, int* token2slot) {
  int t = blockIdx.x * 256 + threadIdx.x;
  if (t >= cT) return;
  #pragma unroll
  for (int j = 0; j < 2; ++j) {
    int e = tokExp[2*t + j];
    int pos = atomicAdd(&cursor[e], 1);
    int slot = padOff[e] + pos;
    slot2token[slot] = t;
    token2slot[2*t + j] = slot;
  }
}

__global__ void k_silu_mul(float* __restrict__ g, const float* __restrict__ u) {
  size_t i = (size_t)blockIdx.x * 256 + threadIdx.x;
  float gv = g[i];
  g[i] = (gv / (1.f + __expf(-gv))) * u[i];
}

__global__ void k_combine(const int* __restrict__ token2slot, const float* __restrict__ tokW,
                          const float* __restrict__ y, float* __restrict__ x) {
  int t = blockIdx.x;
  int s0 = token2slot[2*t], s1 = token2slot[2*t + 1];
  float w0 = tokW[2*t], w1 = tokW[2*t + 1];
  for (int d = threadIdx.x; d < cD; d += 256)
    x[(size_t)t*cD + d] += w0 * y[(size_t)s0*cD + d] + w1 * y[(size_t)s1*cD + d];
}

// ============================================================================
extern "C" void kernel_launch(void* const* d_in, const int* in_sizes, int n_in,
                              void* d_out, int out_size, void* d_ws, size_t ws_size,
                              hipStream_t stream) {
  const int*   ids     = (const int*)  d_in[0];
  const float* emb     = (const float*)d_in[1];
  const float* attn_nw = (const float*)d_in[2];
  const float* wq      = (const float*)d_in[3];
  const float* wk      = (const float*)d_in[4];
  const float* wvw     = (const float*)d_in[5];
  const float* wo      = (const float*)d_in[6];
  const float* moe_nw  = (const float*)d_in[7];
  const float* rw      = (const float*)d_in[8];
  const float* w1      = (const float*)d_in[9];
  const float* w2      = (const float*)d_in[10];
  const float* w3      = (const float*)d_in[11];
  const float* fin_nw  = (const float*)d_in[12];
  float* out = (float*)d_out;

  char* p = (char*)d_ws;
  auto carve = [&](size_t bytes) -> void* {
    void* r = (void*)p;
    p += (bytes + 255) & ~(size_t)255;
    return r;
  };
  float* x    = (float*)carve((size_t)cT * cD * 4);
  float* xn   = (float*)carve((size_t)cT * cD * 4);
  float* qb   = (float*)carve((size_t)cT * cD * 4);
  float* kb   = (float*)carve((size_t)cT * cD * 4);
  float* vb   = (float*)carve((size_t)cT * cD * 4);
  float* att  = (float*)carve((size_t)cT * cD * 4);
  float* gate = (float*)carve((size_t)SLOT_CAP * cHID * 4);
  float* up   = (float*)carve((size_t)SLOT_CAP * cHID * 4);
  float* yb   = (float*)carve((size_t)SLOT_CAP * cD * 4);
  int*   tokExp     = (int*)  carve(cT * 2 * 4);
  float* tokW       = (float*)carve(cT * 2 * 4);
  int*   token2slot = (int*)  carve(cT * 2 * 4);
  int*   slot2token = (int*)  carve(SLOT_CAP * 4);
  int*   tileExp    = (int*)  carve(SLOT_TILES * 4);
  int*   count      = (int*)  carve(8 * 4);
  int*   cursor     = (int*)  carve(8 * 4);
  int*   padOff     = (int*)  carve(16 * 4);
  int*   totTiles   = (int*)  carve(4);

  const dim3 bA(32, 8);
  k_embed<<<cT, 256, 0, stream>>>(ids, emb, x);

  for (int l = 0; l < cL; ++l) {
    // ---- attention ----
    k_rmsnorm<<<cT, 256, 0, stream>>>(x, attn_nw + (size_t)l * cD, xn);
    dim3 gQ(cT / 64, cD / 64);
    k_gemm<false,false,false><<<gQ, 256, 0, stream>>>(xn, wq  + (size_t)l*cD*cD, qb, cD, cD, nullptr, nullptr, 0, nullptr);
    k_gemm<false,false,false><<<gQ, 256, 0, stream>>>(xn, wk  + (size_t)l*cD*cD, kb, cD, cD, nullptr, nullptr, 0, nullptr);
    k_gemm<false,false,false><<<gQ, 256, 0, stream>>>(xn, wvw + (size_t)l*cD*cD, vb, cD, cD, nullptr, nullptr, 0, nullptr);
    k_rope<<<cT, 256, 0, stream>>>(qb, kb);
    dim3 gA(cT / 8, cH);
    k_attn<<<gA, bA, 0, stream>>>(qb, kb, vb, att);
    k_gemm<false,false,true ><<<gQ, 256, 0, stream>>>(att, wo + (size_t)l*cD*cD, x, cD, cD, nullptr, nullptr, 0, nullptr);

    // ---- MoE: route (top-2), gather-GEMM per expert segment ----
    k_rmsnorm<<<cT, 256, 0, stream>>>(x, moe_nw + (size_t)l * cD, xn);
    k_zero8<<<1, 32, 0, stream>>>(count);
    k_router<<<dim3(cT / 8), bA, 0, stream>>>(xn, rw + (size_t)l*cD*cE, tokExp, tokW, count);
    k_scan_init<<<1, 256, 0, stream>>>(count, padOff, cursor, slot2token, tileExp, totTiles);
    k_assign<<<cT / 256, 256, 0, stream>>>(tokExp, padOff, cursor, slot2token, token2slot);

    const size_t es13 = (size_t)cD * cHID;
    const size_t es2  = (size_t)cHID * cD;
    dim3 gG(SLOT_TILES, cHID / 64);
    k_gemm<true ,false,false><<<gG, 256, 0, stream>>>(xn, w1 + (size_t)l*cE*es13, gate, cHID, cD, slot2token, tileExp, es13, totTiles);
    k_gemm<true ,false,false><<<gG, 256, 0, stream>>>(xn, w3 + (size_t)l*cE*es13, up,   cHID, cD, slot2token, tileExp, es13, totTiles);
    k_silu_mul<<<(SLOT_CAP * cHID) / 256, 256, 0, stream>>>(gate, up);
    dim3 gDn(SLOT_TILES, cD / 64);
    k_gemm<false,false,false><<<gDn, 256, 0, stream>>>(gate, w2 + (size_t)l*cE*es2, yb, cD, cHID, nullptr, tileExp, es2, totTiles);
    k_combine<<<cT, 256, 0, stream>>>(token2slot, tokW, yb, x);
  }

  // ---- final norm + logits (B transposed: tok_embed is [V, D]) ----
  k_rmsnorm<<<cT, 256, 0, stream>>>(x, fin_nw, xn);
  dim3 gL(cT / 64, cV / 64);
  k_gemm<false,true ,false><<<gL, 256, 0, stream>>>(xn, emb, out, cV, cD, nullptr, nullptr, 0, nullptr);
}
